// SimpleCorrector_7352984011301
// MI455X (gfx1250) — matrix operand order repaired and checked by reading, not compiled, not run on hardware
//
#include <hip/hip_runtime.h>
#include <hip/hip_bf16.h>

#define N_NODES 100000
#define N_EDGES 1250000
#define IN_DIM  64
#define HID     128
#define OUT_DIM 64

typedef __attribute__((ext_vector_type(16))) _Float16 v16h;
typedef __attribute__((ext_vector_type(8)))  _Float16 v8h;
typedef __attribute__((ext_vector_type(8)))  float    v8f;

// ---------------------------------------------------------------------------
// Kernel 1: zero the agg+deg scratch (contiguous region)
// ---------------------------------------------------------------------------
__global__ void gnn_zero_kernel(float* __restrict__ p, int n) {
    int i = blockIdx.x * blockDim.x + threadIdx.x;
    if (i < n) p[i] = 0.0f;
}

// ---------------------------------------------------------------------------
// Kernel 2: edge scatter-add. One wave (32 lanes) per edge; each lane handles
// 2 of the 64 features (coalesced 256B row load), atomicAdd into agg.
// ---------------------------------------------------------------------------
__global__ void gnn_scatter_kernel(const float* __restrict__ x,
                                   const long long* __restrict__ ei,
                                   float* __restrict__ agg,
                                   float* __restrict__ deg) {
    int tid  = blockIdx.x * blockDim.x + threadIdx.x;
    int edge = tid >> 5;
    int lane = tid & 31;
    if (edge >= N_EDGES) return;
    int r = (int)ei[edge];             // dst node
    int c = (int)ei[N_EDGES + edge];   // src node
    const float* src = x + (size_t)c * IN_DIM + lane * 2;
    float v0 = src[0];
    float v1 = src[1];
    float* dst = agg + (size_t)r * IN_DIM + lane * 2;
    atomicAdd(dst + 0, v0);
    atomicAdd(dst + 1, v1);
    if (lane == 0) atomicAdd(deg + r, 1.0f);
}

// ---------------------------------------------------------------------------
// Kernel 3: h0 = f16(concat(x, agg / max(deg,1)))  row-major [N_NODES][128]
// ---------------------------------------------------------------------------
__global__ void gnn_finalize_kernel(const float* __restrict__ x,
                                    const float* __restrict__ agg,
                                    const float* __restrict__ deg,
                                    _Float16* __restrict__ h0) {
    int tid = blockIdx.x * blockDim.x + threadIdx.x;
    if (tid >= N_NODES * (2 * IN_DIM)) return;
    int node = tid >> 7;
    int j    = tid & 127;
    float v;
    if (j < IN_DIM) {
        v = x[(size_t)node * IN_DIM + j];
    } else {
        float d = deg[node];
        d = d > 1.0f ? d : 1.0f;
        v = agg[(size_t)node * IN_DIM + (j - IN_DIM)] / d;
    }
    h0[tid] = (_Float16)v;
}

// ---------------------------------------------------------------------------
// Kernel 4: convert weights f32 -> f16, transposed to Wt[n][k] so a B-fragment
// lane (col n = lane&15, K = kt*32 + (lane>>4)*8 + [0..7] and +16) reads two contiguous 16B runs.
// ---------------------------------------------------------------------------
__global__ void gnn_wconv_kernel(const float* __restrict__ W1,
                                 const float* __restrict__ W2,
                                 const float* __restrict__ W3,
                                 const float* __restrict__ W4,
                                 _Float16* __restrict__ w1t,
                                 _Float16* __restrict__ w2t,
                                 _Float16* __restrict__ w3t,
                                 _Float16* __restrict__ w4t) {
    int tid = blockIdx.x * blockDim.x + threadIdx.x;
    if (tid < HID * HID) {
        int k = tid / HID, n = tid % HID;   // source is row-major [K][N]
        w1t[n * HID + k] = (_Float16)W1[tid];
        w2t[n * HID + k] = (_Float16)W2[tid];
        w3t[n * HID + k] = (_Float16)W3[tid];
    }
    if (tid < HID * OUT_DIM) {
        int k = tid / OUT_DIM, n = tid % OUT_DIM;
        w4t[n * HID + k] = (_Float16)W4[tid];
    }
}

// ---------------------------------------------------------------------------
// Kernel 5: fused 4-layer MLP with v_wmma_f32_16x16x32_f16, M-blocked by 2.
// Block = 256 threads = 8 waves. Dynamic LDS:
//   sW1,sW2,sW3: 128x128 f16; sW4: 64x128 f16; sB: 448 f32;
//   sH: 8 waves x (32x128) f16 staging.  Total 182,016 B (< 320KB/WG).
// Each wave processes a 32-row tile: both 16-row A-fragment sets live in
// registers; each B fragment loaded from LDS feeds TWO chained WMMAs.
// ---------------------------------------------------------------------------
__global__ __launch_bounds__(256)
void gnn_mlp_wmma(const _Float16* __restrict__ h0,
                  const _Float16* __restrict__ w1t,
                  const _Float16* __restrict__ w2t,
                  const _Float16* __restrict__ w3t,
                  const _Float16* __restrict__ w4t,
                  const float* __restrict__ b1, const float* __restrict__ b2,
                  const float* __restrict__ b3, const float* __restrict__ b4,
                  float* __restrict__ out) {
    extern __shared__ char smem[];
    _Float16* sW1 = (_Float16*)smem;
    _Float16* sW2 = sW1 + HID * HID;
    _Float16* sW3 = sW2 + HID * HID;
    _Float16* sW4 = sW3 + HID * HID;
    float*    sB  = (float*)(sW4 + OUT_DIM * HID);
    _Float16* sH  = (_Float16*)(sB + 3 * HID + OUT_DIM);

    const int tid = threadIdx.x;

    // Cooperative 16B-vector weight preload into LDS
    {
        int4*       dw;
        const int4* sw;
        dw = (int4*)sW1; sw = (const int4*)w1t;
        for (int i = tid; i < (HID * HID) / 8; i += 256) dw[i] = sw[i];
        dw = (int4*)sW2; sw = (const int4*)w2t;
        for (int i = tid; i < (HID * HID) / 8; i += 256) dw[i] = sw[i];
        dw = (int4*)sW3; sw = (const int4*)w3t;
        for (int i = tid; i < (HID * HID) / 8; i += 256) dw[i] = sw[i];
        dw = (int4*)sW4; sw = (const int4*)w4t;
        for (int i = tid; i < (OUT_DIM * HID) / 8; i += 256) dw[i] = sw[i];
        for (int i = tid; i < HID; i += 256) {
            sB[i]           = b1[i];
            sB[HID + i]     = b2[i];
            sB[2 * HID + i] = b3[i];
        }
        if (tid < OUT_DIM) sB[3 * HID + tid] = b4[tid];
    }
    __syncthreads();

    const int lane = tid & 31;
    const int wave = tid >> 5;
    const int m  = lane & 15;   // A row / C column within the tile
    const int kh = lane >> 4;   // half-wave selector in fragment layouts
    _Float16* myH = sH + wave * (32 * HID);

    const int wgid    = blockIdx.x * 8 + wave;
    const int nwaves  = gridDim.x * 8;
    const int n_tiles = N_NODES / 32;   // 3125, exact

    for (int tile = wgid; tile < n_tiles; tile += nwaves) {
        const int row0 = tile * 32;
        v16h A[2][4];

        // ---- Layers 1..3 (K=128, N=128, bias+ReLU, stage f16 in LDS) ----
        #pragma unroll
        for (int layer = 0; layer < 3; ++layer) {
            // A fragments, 16-bit 16x32 layout: lane holds two 8-elem runs:
            //   K = kt*32 + kh*8 + [0..7]  and  K = kt*32 + 16 + kh*8 + [0..7]
            #pragma unroll
            for (int mt = 0; mt < 2; ++mt) {
                const _Float16* src = (layer == 0)
                    ? (h0 + (size_t)(row0 + mt * 16 + m) * HID)
                    : (myH + (mt * 16 + m) * HID);
                #pragma unroll
                for (int kt = 0; kt < 4; ++kt) {
                    v8h lo = *(const v8h*)(src + kt * 32 + kh * 8);
                    v8h hi = *(const v8h*)(src + kt * 32 + 16 + kh * 8);
                    v16h a;
                    #pragma unroll
                    for (int j = 0; j < 8; ++j) { a[j] = lo[j]; a[j + 8] = hi[j]; }
                    A[mt][kt] = a;
                }
            }
            const _Float16* W = (layer == 0) ? sW1 : (layer == 1 ? sW2 : sW3);
            const float* bias = sB + layer * HID;
            #pragma unroll
            for (int nt = 0; nt < 8; ++nt) {
                const float bv = bias[nt * 16 + m];
                v8f acc0 = {bv, bv, bv, bv, bv, bv, bv, bv};
                v8f acc1 = acc0;
                #pragma unroll
                for (int kt = 0; kt < 4; ++kt) {
                    // B fragment, 32x16: lane col n=m, K = kt*32 + kh*8 + [0..7] and kt*32 + 16 + kh*8 + [0..7]
                    const _Float16* wp = W + (nt * 16 + m) * HID + kt * 32 + kh * 8;
                    v8h blo = *(const v8h*)(wp);
                    v8h bhi = *(const v8h*)(wp + 16);
                    v16h b;
                    #pragma unroll
                    for (int j = 0; j < 8; ++j) { b[j] = blo[j]; b[j + 8] = bhi[j]; }
                    acc0 = __builtin_amdgcn_wmma_f32_16x16x32_f16(
                        false, A[0][kt], false, b, (short)0, acc0, false, false);
                    acc1 = __builtin_amdgcn_wmma_f32_16x16x32_f16(
                        false, A[1][kt], false, b, (short)0, acc1, false, false);
                }
                // C layout: lane col n=m, row = kh*8 + r for VGPR r
                #pragma unroll
                for (int r = 0; r < 8; ++r) {
                    float v0 = acc0[r];
                    float v1 = acc1[r];
                    v0 = v0 > 0.0f ? v0 : 0.0f;
                    v1 = v1 > 0.0f ? v1 : 0.0f;
                    myH[(kh * 8 + r) * HID + nt * 16 + m]        = (_Float16)v0;
                    myH[(16 + kh * 8 + r) * HID + nt * 16 + m]   = (_Float16)v1;
                }
            }
        }

        // ---- Layer 4 (K=128, N=64, bias, f32 store to out) ----
        {
            #pragma unroll
            for (int mt = 0; mt < 2; ++mt) {
                const _Float16* src = myH + (mt * 16 + m) * HID;
                #pragma unroll
                for (int kt = 0; kt < 4; ++kt) {
                    v8h lo = *(const v8h*)(src + kt * 32 + kh * 8);
                    v8h hi = *(const v8h*)(src + kt * 32 + 16 + kh * 8);
                    v16h a;
                    #pragma unroll
                    for (int j = 0; j < 8; ++j) { a[j] = lo[j]; a[j + 8] = hi[j]; }
                    A[mt][kt] = a;
                }
            }
            #pragma unroll
            for (int nt = 0; nt < 4; ++nt) {
                const float bv = sB[3 * HID + nt * 16 + m];
                v8f acc0 = {bv, bv, bv, bv, bv, bv, bv, bv};
                v8f acc1 = acc0;
                #pragma unroll
                for (int kt = 0; kt < 4; ++kt) {
                    const _Float16* wp = sW4 + (nt * 16 + m) * HID + kt * 32 + kh * 8;
                    v8h blo = *(const v8h*)(wp);
                    v8h bhi = *(const v8h*)(wp + 16);
                    v16h b;
                    #pragma unroll
                    for (int j = 0; j < 8; ++j) { b[j] = blo[j]; b[j + 8] = bhi[j]; }
                    acc0 = __builtin_amdgcn_wmma_f32_16x16x32_f16(
                        false, A[0][kt], false, b, (short)0, acc0, false, false);
                    acc1 = __builtin_amdgcn_wmma_f32_16x16x32_f16(
                        false, A[1][kt], false, b, (short)0, acc1, false, false);
                }
                #pragma unroll
                for (int r = 0; r < 8; ++r) {
                    out[(size_t)(row0 + kh * 8 + r) * OUT_DIM + nt * 16 + m]      = acc0[r];
                    out[(size_t)(row0 + 16 + kh * 8 + r) * OUT_DIM + nt * 16 + m] = acc1[r];
                }
            }
        }
    }
}

// ---------------------------------------------------------------------------
// Launch
// ---------------------------------------------------------------------------
extern "C" void kernel_launch(void* const* d_in, const int* in_sizes, int n_in,
                              void* d_out, int out_size, void* d_ws, size_t ws_size,
                              hipStream_t stream) {
    (void)in_sizes; (void)n_in; (void)out_size; (void)ws_size;
    const float*     x  = (const float*)d_in[0];
    const long long* ei = (const long long*)d_in[1];
    const float* W1 = (const float*)d_in[2];
    const float* b1 = (const float*)d_in[3];
    const float* W2 = (const float*)d_in[4];
    const float* b2 = (const float*)d_in[5];
    const float* W3 = (const float*)d_in[6];
    const float* b3 = (const float*)d_in[7];
    const float* W4 = (const float*)d_in[8];
    const float* b4 = (const float*)d_in[9];
    float* out = (float*)d_out;

    char* ws = (char*)d_ws;
    size_t off = 0;
    float* agg = (float*)(ws + off); off += (size_t)N_NODES * IN_DIM * sizeof(float);
    float* deg = (float*)(ws + off); off += (size_t)N_NODES * sizeof(float);
    _Float16* h0  = (_Float16*)(ws + off); off += (size_t)N_NODES * (2 * IN_DIM) * sizeof(_Float16);
    _Float16* w1t = (_Float16*)(ws + off); off += (size_t)HID * HID * sizeof(_Float16);
    _Float16* w2t = (_Float16*)(ws + off); off += (size_t)HID * HID * sizeof(_Float16);
    _Float16* w3t = (_Float16*)(ws + off); off += (size_t)HID * HID * sizeof(_Float16);
    _Float16* w4t = (_Float16*)(ws + off); off += (size_t)OUT_DIM * HID * sizeof(_Float16);

    // 1) zero agg + deg (contiguous)
    {
        int n = N_NODES * (IN_DIM + 1);
        gnn_zero_kernel<<<(n + 255) / 256, 256, 0, stream>>>(agg, n);
    }
    // 2) edge scatter (1 wave per edge)
    {
        long long total = (long long)N_EDGES * 32;
        int blocks = (int)((total + 255) / 256);
        gnn_scatter_kernel<<<blocks, 256, 0, stream>>>(x, ei, agg, deg);
    }
    // 3) finalize h0
    {
        int n = N_NODES * (2 * IN_DIM);
        gnn_finalize_kernel<<<(n + 255) / 256, 256, 0, stream>>>(x, agg, deg, h0);
    }
    // 4) weight convert/transpose
    gnn_wconv_kernel<<<(HID * HID + 255) / 256, 256, 0, stream>>>(
        W1, W2, W3, W4, w1t, w2t, w3t, w4t);
    // 5) fused WMMA MLP (M-blocked by 2)
    {
        size_t smem = (size_t)(3 * HID * HID + OUT_DIM * HID) * sizeof(_Float16)
                    + (size_t)(3 * HID + OUT_DIM) * sizeof(float)
                    + (size_t)8 * 32 * HID * sizeof(_Float16);   // 182,016 B
        (void)hipFuncSetAttribute((const void*)gnn_mlp_wmma,
                                  hipFuncAttributeMaxDynamicSharedMemorySize,
                                  (int)smem);
        gnn_mlp_wmma<<<256, 256, smem, stream>>>(h0, w1t, w2t, w3t, w4t,
                                                 b1, b2, b3, b4, out);
    }
}
